// GRULayer_32195074851538
// MI455X (gfx1250) — matrix operand order repaired and checked by reading, not compiled, not run on hardware
//
#include <hip/hip_runtime.h>

// ---------------- problem constants (match reference) ----------------
#define S_LEN 1024
#define BATCH 64
#define IDIM  1024
#define HDIM  1024
#define KC    128                 // K chunk staged in LDS
#define LDS_PITCH (KC + 8)        // pad: rows start 272B apart (16B aligned, bank-spread)
#define NWG   32                  // persistent workgroups (32 cols each)
#define COLS_PER_WG 32

#define AS1 __attribute__((address_space(1)))
#define AS3 __attribute__((address_space(3)))

typedef __attribute__((ext_vector_type(16))) _Float16 v16h;
typedef __attribute__((ext_vector_type(8)))  _Float16 v8h;
typedef __attribute__((ext_vector_type(4)))  _Float16 v4h;
typedef __attribute__((ext_vector_type(8)))  float    v8f;
typedef int i4v __attribute__((__vector_size__(4 * sizeof(int))));   // b128 payload type per builtin signature

union V16 { v16h v; v8h h[2]; };

// ---------------- CDNA5 async global->LDS copy (ASYNCcnt path) ----------------
#if defined(__has_builtin)
# if __has_builtin(__builtin_amdgcn_global_load_async_to_lds_b128)
#  define HAVE_ASYNC_COPY 1
# endif
#endif
#ifndef HAVE_ASYNC_COPY
# define HAVE_ASYNC_COPY 0
#endif

__device__ __forceinline__ void copy16B_g2l(const _Float16* g, _Float16* l) {
#if HAVE_ASYNC_COPY
  __builtin_amdgcn_global_load_async_to_lds_b128((AS1 i4v*)g, (AS3 i4v*)l, 0, 0);
#else
  *(v8h*)l = *(const v8h*)g;
#endif
}

__device__ __forceinline__ void copy_wait() {
#if HAVE_ASYNC_COPY
# if __has_builtin(__builtin_amdgcn_s_wait_asynccnt)
  __builtin_amdgcn_s_wait_asynccnt(0);
# else
  asm volatile("s_wait_asynccnt 0" ::: "memory");
# endif
#endif
}

// ---------------- helper kernels ----------------
__global__ void cvt_f32_to_f16(const float* __restrict__ src, _Float16* __restrict__ dst, int n) {
  int i = blockIdx.x * blockDim.x + threadIdx.x;
  if (i < n) dst[i] = (_Float16)src[i];
}

__global__ void init_state(float* hF32, _Float16* hF16, unsigned* counter) {
  int i = blockIdx.x * blockDim.x + threadIdx.x;
  if (i < BATCH * HDIM) { hF32[i] = 0.f; hF16[i] = (_Float16)0.f; }
  if (i == 0) *counter = 0u;
}

// ---------------- fragment loads (per CDNA5 ISA VGPR layouts) ----------------
// A 16x32 f16: lane<16 holds M=lane, K = {k..k+7, k+16..k+23}; lane>=16 holds K = {k+8..k+15, k+24..k+31}
__device__ __forceinline__ v16h load_a_frag(const _Float16* Abuf, int mbase, int k, int lane) {
  const int m = mbase + (lane & 15);
  const int halfId = lane >> 4;
  const _Float16* p = Abuf + m * LDS_PITCH + k + halfId * 8;
  V16 f;
  f.h[0] = *(const v8h*)p;          // K = k + half*8 .. +7
  f.h[1] = *(const v8h*)(p + 16);   // K = k + 16 + half*8 .. +7
  return f.v;
}

// B 32x16 f16 for C = A * W^T: B[k][n] = W[n][k]; lane<16: col n=lane, K = {k..k+7, k+16..k+23}; lane>=16: K = {k+8..k+15, k+24..k+31}
__device__ __forceinline__ v16h load_b_frag(const _Float16* __restrict__ W, int col0, int kg, int lane) {
  const int n = lane & 15;
  const int halfId = lane >> 4;
  const _Float16* p = W + (size_t)(col0 + n) * IDIM + kg + halfId * 8;
  V16 f;
  f.h[0] = *(const v8h*)p;
  f.h[1] = *(const v8h*)(p + 16);
  return f.v;
}

// prefetch the next K-chunk of this lane's B-fragment rows (emits global_prefetch_b8)
__device__ __forceinline__ void prefetch_b(const _Float16* __restrict__ W, int col0, int kg, int lane) {
  const int n = lane & 15;
  const int halfId = lane >> 4;
  __builtin_prefetch(W + (size_t)(col0 + n) * IDIM + kg + halfId * 16, 0, 3);
}

__device__ __forceinline__ v8f wmma_f16(v16h a, v16h b, v8f c) {
  return __builtin_amdgcn_wmma_f32_16x16x32_f16(false, a, false, b, (short)0, c, false, false);
}

// ---------------- grid-wide barrier (monotonic counter, sense-free) ----------------
__device__ __forceinline__ void grid_barrier(unsigned* counter, unsigned target) {
  __threadfence();                 // release this block's global writes
  __syncthreads();
  if (threadIdx.x == 0) {
    __hip_atomic_fetch_add(counter, 1u, __ATOMIC_RELEASE, __HIP_MEMORY_SCOPE_AGENT);
    while (__hip_atomic_load(counter, __ATOMIC_ACQUIRE, __HIP_MEMORY_SCOPE_AGENT) < target)
      __builtin_amdgcn_s_sleep(1);
  }
  __syncthreads();
  __threadfence();                 // acquire other blocks' writes
}

__device__ __forceinline__ float sigmoidf_(float x) { return 1.f / (1.f + __expf(-x)); }

// ---------------- persistent GRU kernel ----------------
__global__ __launch_bounds__(256, 1)
void gru_persistent(const float* __restrict__ x,
                    const _Float16* __restrict__ Wihr, const _Float16* __restrict__ Whhr,
                    const _Float16* __restrict__ Wihz, const _Float16* __restrict__ Whhz,
                    const _Float16* __restrict__ Wihn, const _Float16* __restrict__ Whhn,
                    const float* __restrict__ b_ih_r, const float* __restrict__ b_hh_r,
                    const float* __restrict__ b_ih_z, const float* __restrict__ b_hh_z,
                    const float* __restrict__ b_ih_n, const float* __restrict__ b_hh_n,
                    float* __restrict__ out,
                    float* __restrict__ hF32, _Float16* __restrict__ hF16,
                    _Float16* __restrict__ rhF16, unsigned* __restrict__ counter)
{
  __shared__ _Float16 Ax[BATCH * LDS_PITCH];   // x_t chunk (f16)
  __shared__ _Float16 Ah[BATCH * LDS_PITCH];   // h chunk (f16)

  const int tid  = threadIdx.x;
  const int wave = tid >> 5;
  const int lane = tid & 31;
  const int mi = wave & 3;                 // row block  (4 x 16 rows = 64)
  const int ni = wave >> 2;                // col block  (2 x 16 cols = 32)
  const int mbase = mi * 16;
  const int col0  = blockIdx.x * COLS_PER_WG + ni * 16;
  const int n_    = lane & 15;
  const int halfId = lane >> 4;
  const int j     = col0 + n_;             // this lane's output column
  const int mrow0 = mbase + 8 * halfId;    // first of 8 rows this lane holds (C layout)

  const float bias_r = b_ih_r[j] + b_hh_r[j];
  const float bias_z = b_ih_z[j] + b_hh_z[j];
  const float bias_n = b_ih_n[j] + b_hh_n[j];

  #pragma unroll 1
  for (int s = 0; s < S_LEN; ++s) {
    const float* xs = x + (size_t)s * BATCH * IDIM;
    v8f accR = {}; v8f accZ = {}; v8f accN = {};

    // ---- phase 1: accR = x Wr^T + h Whr^T ; accZ likewise ; accN = x Wn^T ----
    #pragma unroll 1
    for (int kc = 0; kc < IDIM; kc += KC) {
      // async-stage h chunk (ASYNCcnt path on CDNA5)
      for (int i = tid; i < BATCH * (KC / 8); i += 256) {
        int row = i / (KC / 8), cv = i % (KC / 8);
        copy16B_g2l(hF16 + (size_t)row * HDIM + kc + cv * 8,
                    Ah + row * LDS_PITCH + cv * 8);
      }
      // x f32 -> f16 into LDS (needs conversion: VGPR path) + prefetch next chunk
      for (int i = tid; i < BATCH * (KC / 4); i += 256) {
        int row = i / (KC / 4), cv = i % (KC / 4);
        const float* gp = xs + (size_t)row * IDIM + kc;
        float4 v = ((const float4*)gp)[cv];
        if (kc + KC < IDIM) __builtin_prefetch(gp + KC + cv * 4, 0, 0);
        v4h t = { (_Float16)v.x, (_Float16)v.y, (_Float16)v.z, (_Float16)v.w };
        *(v4h*)(Ax + row * LDS_PITCH + cv * 4) = t;
      }
      // prefetch next chunk of the weight streams for this lane's B rows
      if (kc + KC < IDIM) {
        const int kpf = kc + KC;
        prefetch_b(Wihr, col0, kpf, lane);
        prefetch_b(Whhr, col0, kpf, lane);
        prefetch_b(Wihz, col0, kpf, lane);
        prefetch_b(Whhz, col0, kpf, lane);
        prefetch_b(Wihn, col0, kpf, lane);
      }
      copy_wait();
      __syncthreads();
      #pragma unroll
      for (int k = 0; k < KC; k += 32) {
        v16h aX = load_a_frag(Ax, mbase, k, lane);
        v16h aH = load_a_frag(Ah, mbase, k, lane);
        const int kg = kc + k;
        accR = wmma_f16(aX, load_b_frag(Wihr, col0, kg, lane), accR);
        accR = wmma_f16(aH, load_b_frag(Whhr, col0, kg, lane), accR);
        accZ = wmma_f16(aX, load_b_frag(Wihz, col0, kg, lane), accZ);
        accZ = wmma_f16(aH, load_b_frag(Whhz, col0, kg, lane), accZ);
        accN = wmma_f16(aX, load_b_frag(Wihn, col0, kg, lane), accN);
      }
      __syncthreads();
    }

    // ---- phase 2: r = sigmoid(...), publish r*h ; keep z ----
    float zv[8];
    #pragma unroll
    for (int r = 0; r < 8; ++r) {
      const int m = mrow0 + r;
      float rv = sigmoidf_(accR[r] + bias_r);
      float hv = hF32[m * HDIM + j];
      rhF16[m * HDIM + j] = (_Float16)(rv * hv);
      zv[r] = sigmoidf_(accZ[r] + bias_z);
    }

    grid_barrier(counter, (unsigned)(2 * s + 1) * NWG);

    // ---- phase 3: accN += (r*h) Whn^T ----
    #pragma unroll 1
    for (int kc = 0; kc < HDIM; kc += KC) {
      for (int i = tid; i < BATCH * (KC / 8); i += 256) {
        int row = i / (KC / 8), cv = i % (KC / 8);
        copy16B_g2l(rhF16 + (size_t)row * HDIM + kc + cv * 8,
                    Ax + row * LDS_PITCH + cv * 8);
      }
      if (kc + KC < HDIM) prefetch_b(Whhn, col0, kc + KC, lane);
      copy_wait();
      __syncthreads();
      #pragma unroll
      for (int k = 0; k < KC; k += 32) {
        v16h aR = load_a_frag(Ax, mbase, k, lane);
        accN = wmma_f16(aR, load_b_frag(Whhn, col0, kc + k, lane), accN);
      }
      __syncthreads();
    }

    // ---- phase 4: gate combine, write output + new h ----
    float* outs = out + (size_t)s * BATCH * HDIM;
    #pragma unroll
    for (int r = 0; r < 8; ++r) {
      const int m = mrow0 + r;
      float nv = tanhf(accN[r] + bias_n);
      float hv = hF32[m * HDIM + j];
      float hn = (1.f - zv[r]) * nv + zv[r] * hv;
      outs[m * HDIM + j] = hn;
      hF32[m * HDIM + j] = hn;
      hF16[m * HDIM + j] = (_Float16)hn;
      if (s == S_LEN - 1) out[(size_t)S_LEN * BATCH * HDIM + m * HDIM + j] = hn;  // h_n
    }

    grid_barrier(counter, (unsigned)(2 * s + 2) * NWG);
  }
}

// ---------------- host launch ----------------
extern "C" void kernel_launch(void* const* d_in, const int* in_sizes, int n_in,
                              void* d_out, int out_size, void* d_ws, size_t ws_size,
                              hipStream_t stream) {
  const float* x      = (const float*)d_in[0];
  const float* wihr   = (const float*)d_in[1];
  const float* whhr   = (const float*)d_in[2];
  const float* wihz   = (const float*)d_in[3];
  const float* whhz   = (const float*)d_in[4];
  const float* wihn   = (const float*)d_in[5];
  const float* whhn   = (const float*)d_in[6];
  const float* b_ih_r = (const float*)d_in[7];
  const float* b_hh_r = (const float*)d_in[8];
  const float* b_ih_z = (const float*)d_in[9];
  const float* b_hh_z = (const float*)d_in[10];
  const float* b_ih_n = (const float*)d_in[11];
  const float* b_hh_n = (const float*)d_in[12];
  float* out = (float*)d_out;

  // workspace layout (~12.5 MB total)
  char* ws = (char*)d_ws;
  const size_t WBYTES = (size_t)HDIM * IDIM * sizeof(_Float16);   // 2 MB each
  _Float16* Wihr16 = (_Float16*)(ws + 0 * WBYTES);
  _Float16* Whhr16 = (_Float16*)(ws + 1 * WBYTES);
  _Float16* Wihz16 = (_Float16*)(ws + 2 * WBYTES);
  _Float16* Whhz16 = (_Float16*)(ws + 3 * WBYTES);
  _Float16* Wihn16 = (_Float16*)(ws + 4 * WBYTES);
  _Float16* Whhn16 = (_Float16*)(ws + 5 * WBYTES);
  char* p = ws + 6 * WBYTES;
  float*    hF32  = (float*)p;            p += (size_t)BATCH * HDIM * sizeof(float);
  _Float16* hF16  = (_Float16*)p;         p += (size_t)BATCH * HDIM * sizeof(_Float16);
  _Float16* rhF16 = (_Float16*)p;         p += (size_t)BATCH * HDIM * sizeof(_Float16);
  unsigned* counter = (unsigned*)p;

  const int nW = HDIM * IDIM;
  const int cb = (nW + 255) / 256;
  cvt_f32_to_f16<<<cb, 256, 0, stream>>>(wihr, Wihr16, nW);
  cvt_f32_to_f16<<<cb, 256, 0, stream>>>(whhr, Whhr16, nW);
  cvt_f32_to_f16<<<cb, 256, 0, stream>>>(wihz, Wihz16, nW);
  cvt_f32_to_f16<<<cb, 256, 0, stream>>>(whhz, Whhz16, nW);
  cvt_f32_to_f16<<<cb, 256, 0, stream>>>(wihn, Wihn16, nW);
  cvt_f32_to_f16<<<cb, 256, 0, stream>>>(whhn, Whhn16, nW);
  init_state<<<(BATCH * HDIM + 255) / 256, 256, 0, stream>>>(hF32, hF16, counter);

  gru_persistent<<<NWG, 256, 0, stream>>>(x,
      Wihr16, Whhr16, Wihz16, Whhz16, Wihn16, Whhn16,
      b_ih_r, b_hh_r, b_ih_z, b_hh_z, b_ih_n, b_hh_n,
      out, hF32, hF16, rhF16, counter);
}
